// Attention_38190849196482
// MI455X (gfx1250) — matrix operand order repaired and checked by reading, not compiled, not run on hardware
//
#include <hip/hip_runtime.h>

// ============================================================================
// MLA (DeepSeek-style) forward for MI455X (gfx1250): wave32 + WMMA bf16.
// - All matmuls on v_wmma_f32_16x16x32_bf16 (fp32 accumulate).
// - V is stored transposed [B,NH,D,S] at pack time so the P*V B-fragment is a
//   contiguous 32B global load per lane (no LDS V staging at all).
// - Only the 16x32 P tile round-trips LDS (D-layout -> A-layout transpose),
//   fenced with the CDNA5 split counter s_wait_dscnt.
// - GEMM: 32x64 tile per wave (8 WMMA per 32-K step) for 2x arithmetic
//   intensity vs a 16-row tile; 128x64 per 4-wave workgroup.
// ============================================================================

typedef __bf16 bf16_t;
typedef __attribute__((ext_vector_type(8)))  __bf16 v8bf;
typedef __attribute__((ext_vector_type(16))) __bf16 v16bf;
typedef __attribute__((ext_vector_type(8)))  float  v8f;

namespace {
constexpr int kB   = 2;
constexpr int kS   = 2048;
constexpr int kDim = 2048;
constexpr int kNH  = 16;
constexpr int kQL  = 1536;   // Q_LORA
constexpr int kKVL = 512;    // KV_LORA
constexpr int kNope = 128, kRope = 64, kVhd = 128, kQkHd = 192;
constexpr int kM   = kB * kS;             // 4096 token rows
constexpr float kScale = 0.07216878364870323f;  // 192^-0.5
}

// ---------------------------------------------------------------------------
// helpers
// ---------------------------------------------------------------------------
__device__ __forceinline__ v16bf cat8(v8bf lo, v8bf hi) {
  return __builtin_shufflevector(lo, hi, 0, 1, 2, 3, 4, 5, 6, 7,
                                 8, 9, 10, 11, 12, 13, 14, 15);
}

// A-matrix 16x32 bf16 fragment (wave32, contiguous K order matching B):
//   lanes 0-15 : M=lane,    K = {0..15}
//   lanes 16-31: M=lane-16, K = {16..31}
__device__ __forceinline__ v16bf load_a_frag(const bf16_t* rowp, int k, int hf) {
  v8bf lo = *(const v8bf*)(rowp + k + hf * 16);
  v8bf hi = *(const v8bf*)(rowp + k + hf * 16 + 8);
  return cat8(lo, hi);
}

__device__ __forceinline__ v8f wmma_bf16(v16bf a, v16bf b, v8f c) {
  return __builtin_amdgcn_wmma_f32_16x16x32_bf16(
      /*neg_a=*/false, a, /*neg_b=*/false, b,
      /*c_mod=*/(short)0, c, /*reuse_a=*/false, /*reuse_b=*/false);
}

// ---------------------------------------------------------------------------
// f32 -> bf16 convert (element counts here are always multiples of 4)
// ---------------------------------------------------------------------------
__global__ __launch_bounds__(256)
void k_cvt_bf16(const float* __restrict__ in, bf16_t* __restrict__ out, int n) {
  int i = (blockIdx.x * 256 + threadIdx.x) * 4;
  if (i + 3 < n) {
    float4 v = *(const float4*)(in + i);
    out[i + 0] = (bf16_t)v.x;
    out[i + 1] = (bf16_t)v.y;
    out[i + 2] = (bf16_t)v.z;
    out[i + 3] = (bf16_t)v.w;
  } else {
    for (; i < n; ++i) out[i] = (bf16_t)in[i];
  }
}

// ---------------------------------------------------------------------------
// C[M,N] = A[M,K](bf16) @ W[N,K](bf16)^T + bias;  out bf16 and/or f32.
// Workgroup = 4 waves, 128x64 block tile; wave w owns rows [m0, m0+32) x 64
// cols (2 A fragments x 4 W fragments -> 8 WMMAs per 32-K step).
// W fragment: lane n holds W[n0+n][k .. k+15] (contiguous, 32B).
// ---------------------------------------------------------------------------
__global__ __launch_bounds__(128)
void k_gemm_bf16_nt(const bf16_t* __restrict__ A, const bf16_t* __restrict__ W,
                    const float* __restrict__ bias,
                    float* __restrict__ outf, bf16_t* __restrict__ outb,
                    int M, int N, int K) {
  const int lane = threadIdx.x & 31;
  const int wave = threadIdx.x >> 5;
  const int hf   = lane >> 4;   // 0: lanes 0-15, 1: lanes 16-31
  const int l16  = lane & 15;
  const int m0   = blockIdx.y * 128 + wave * 32;
  const int n0   = blockIdx.x * 64;

  v8f acc[2][4] = {};
  const bf16_t* arow0 = A + (size_t)(m0 + l16) * K;
  const bf16_t* arow1 = A + (size_t)(m0 + 16 + l16) * K;

  for (int k = 0; k < K; k += 32) {
    v16bf a0 = load_a_frag(arow0, k, hf);
    v16bf a1 = load_a_frag(arow1, k, hf);
#pragma unroll
    for (int t = 0; t < 4; ++t) {
      const bf16_t* wrow = W + (size_t)(n0 + t * 16 + l16) * K + k + hf * 16;
      v16bf bfrag = *(const v16bf*)wrow;
      acc[0][t] = wmma_bf16(a0, bfrag, acc[0][t]);
      acc[1][t] = wmma_bf16(a1, bfrag, acc[1][t]);
    }
  }

#pragma unroll
  for (int u = 0; u < 2; ++u) {
#pragma unroll
    for (int t = 0; t < 4; ++t) {
      int col = n0 + t * 16 + l16;
      float bv = bias ? bias[col] : 0.f;
#pragma unroll
      for (int r = 0; r < 8; ++r) {
        int row = m0 + u * 16 + r + hf * 8;  // D layout: VGPR r -> M=r (+8 hi)
        float v = acc[u][t][r] + bv;
        if (outf) outf[(size_t)row * N + col] = v;
        if (outb) outb[(size_t)row * N + col] = (bf16_t)v;
      }
    }
  }
}

// ---------------------------------------------------------------------------
// RMSNorm over bf16 rows (f32 statistics): out = in * rsqrt(mean(in^2)+eps) * w
// ---------------------------------------------------------------------------
__global__ __launch_bounds__(256)
void k_rmsnorm(const bf16_t* __restrict__ in, const float* __restrict__ w,
               bf16_t* __restrict__ out, int N, int inStride) {
  const int row = blockIdx.x;
  const bf16_t* r = in + (size_t)row * inStride;
  float ss = 0.f;
  for (int i = threadIdx.x; i < N; i += 256) {
    float v = (float)r[i];
    ss += v * v;
  }
  __shared__ float red[8];
  for (int off = 16; off; off >>= 1) ss += __shfl_xor(ss, off, 32);
  if ((threadIdx.x & 31) == 0) red[threadIdx.x >> 5] = ss;
  __syncthreads();
  if (threadIdx.x < 32) {
    float v = (threadIdx.x < 8) ? red[threadIdx.x] : 0.f;
    for (int off = 4; off; off >>= 1) v += __shfl_xor(v, off, 32);
    if (threadIdx.x == 0) red[0] = v;
  }
  __syncthreads();
  const float inv = rsqrtf(red[0] / (float)N + 1e-6f);
  for (int i = threadIdx.x; i < N; i += 256)
    out[(size_t)row * N + i] = (bf16_t)((float)r[i] * inv * w[i]);
}

// ---------------------------------------------------------------------------
// RoPE + repack to head-major bf16:
//   qh/kh [B,NH,S,192]; V stored TRANSPOSED: vht [B,NH,128,S].
// One block per token row m = b*S + s.
// ---------------------------------------------------------------------------
__global__ __launch_bounds__(256)
void k_pack_rope(const bf16_t* __restrict__ qfull,  // [M, NH*192]
                 const bf16_t* __restrict__ kva,    // [M, 576]  (kv_c | k_pe)
                 const bf16_t* __restrict__ kvb,    // [M, NH*256] (k_nope | v)
                 const float* __restrict__ fc, const float* __restrict__ fs,
                 bf16_t* __restrict__ qh, bf16_t* __restrict__ kh,
                 bf16_t* __restrict__ vht) {
  const int m = blockIdx.x;
  const int b = m / kS, s = m % kS;
  __shared__ float kpe[kRope];

  if (threadIdx.x < kRope / 2) {
    int i = threadIdx.x;
    float xr = (float)kva[(size_t)m * 576 + kKVL + 2 * i];
    float xi = (float)kva[(size_t)m * 576 + kKVL + 2 * i + 1];
    float c = fc[(size_t)s * 32 + i], sn = fs[(size_t)s * 32 + i];
    kpe[2 * i]     = xr * c - xi * sn;
    kpe[2 * i + 1] = xr * sn + xi * c;
  }
  __syncthreads();

  const bf16_t* qrow = qfull + (size_t)m * (kNH * kQkHd);
  for (int idx = threadIdx.x; idx < kNH * kQkHd; idx += 256) {
    int h = idx / kQkHd, d = idx % kQkHd;
    float val;
    if (d < kNope) {
      val = (float)qrow[h * kQkHd + d];
    } else {
      int j = d - kNope, i = j >> 1;
      float xr = (float)qrow[h * kQkHd + kNope + 2 * i];
      float xi = (float)qrow[h * kQkHd + kNope + 2 * i + 1];
      float c = fc[(size_t)s * 32 + i], sn = fs[(size_t)s * 32 + i];
      val = (j & 1) ? (xr * sn + xi * c) : (xr * c - xi * sn);
    }
    qh[(((size_t)b * kNH + h) * kS + s) * kQkHd + d] = (bf16_t)val;
  }

  const bf16_t* kvrow = kvb + (size_t)m * (kNH * 256);
  for (int idx = threadIdx.x; idx < kNH * kQkHd; idx += 256) {
    int h = idx / kQkHd, d = idx % kQkHd;
    float val = (d < kNope) ? (float)kvrow[h * 256 + d] : kpe[d - kNope];
    kh[(((size_t)b * kNH + h) * kS + s) * kQkHd + d] = (bf16_t)val;
  }
  // V transposed: vht[((b*NH+h)*128 + d)*S + s]
  for (int idx = threadIdx.x; idx < kNH * kVhd; idx += 256) {
    int h = idx / kVhd, d = idx % kVhd;
    vht[(((size_t)b * kNH + h) * kVhd + d) * kS + s] = kvrow[h * 256 + kNope + d];
  }
}

// ---------------------------------------------------------------------------
// Flash attention: 1 wave per block; block (qt,h,b) handles 16 query rows,
// streams 32-key tiles. Scores: 12x WMMA (d=192). P*V: 8x WMMA with the V
// fragment loaded contiguously from the transposed global layout. Online
// softmax in fp32; only P transposes through LDS (1KB).
// ---------------------------------------------------------------------------
__global__ __launch_bounds__(32)
void k_mla_attn(const bf16_t* __restrict__ qh, const bf16_t* __restrict__ kh,
                const bf16_t* __restrict__ vht, bf16_t* __restrict__ ao) {
  const int qt = blockIdx.x, h = blockIdx.y, b = blockIdx.z;
  const int lane = threadIdx.x;
  const int hf = lane >> 4, l16 = lane & 15;
  const int qlo = qt * 16;
  const size_t bh = (size_t)b * kNH + h;
  const bf16_t* Q  = qh  + (bh * kS + qlo) * kQkHd;
  const bf16_t* Kb = kh  + bh * kS * kQkHd;
  const bf16_t* Vt = vht + bh * (size_t)kVhd * kS;   // [128][S]

  __shared__ bf16_t pt[16][32];    // P tile, row-major: pt[qrow][key]

  // Q fragments for all 6 K=32 chunks of the 192-d head (held in VGPRs)
  v16bf qf[6];
  {
    const bf16_t* qrow = Q + (size_t)l16 * kQkHd;
#pragma unroll
    for (int c = 0; c < 6; ++c) qf[c] = load_a_frag(qrow, c * 32, hf);
  }

  v8f acc[8] = {};            // 16x128 output accumulator (fp32)
  float mr[8], lr[8];
#pragma unroll
  for (int r = 0; r < 8; ++r) { mr[r] = -INFINITY; lr[r] = 0.f; }

  const int qhi = qlo + 15;
  for (int kt = 0; kt <= qhi; kt += 32) {
    if (kt + 32 <= qhi) {     // prefetch next K/V tiles (global_prefetch_b8)
      __builtin_prefetch(Kb + (size_t)(kt + 32 + l16) * kQkHd, 0, 1);
      __builtin_prefetch(Vt + (size_t)(l16 * 8) * kS + kt + 32, 0, 1);
    }

    // ---- scores: two 16x16 tiles (keys kt..kt+15 and kt+16..kt+31) ----
    v8f s0 = {}, s1 = {};
    const int key0 = kt + l16;
    const int key1 = kt + 16 + l16;
    const int k1c  = (key1 < kS) ? key1 : (kS - 1);  // clamp load, mask later
#pragma unroll
    for (int c = 0; c < 6; ++c) {
      v16bf kf0 = *(const v16bf*)(Kb + (size_t)key0 * kQkHd + c * 32 + hf * 16);
      s0 = wmma_bf16(qf[c], kf0, s0);
      v16bf kf1 = *(const v16bf*)(Kb + (size_t)k1c * kQkHd + c * 32 + hf * 16);
      s1 = wmma_bf16(qf[c], kf1, s1);
    }

    // ---- online softmax (rows r+8*hf, cols = lane within 16-group) ----
#pragma unroll
    for (int r = 0; r < 8; ++r) {
      const int row = qlo + r + hf * 8;
      float a0 = (key0 <= row) ? s0[r] * kScale : -INFINITY;
      float a1 = (key1 <= row) ? s1[r] * kScale : -INFINITY;
      float mx = fmaxf(a0, a1);
      mx = fmaxf(mx, __shfl_xor(mx, 1, 32));
      mx = fmaxf(mx, __shfl_xor(mx, 2, 32));
      mx = fmaxf(mx, __shfl_xor(mx, 4, 32));
      mx = fmaxf(mx, __shfl_xor(mx, 8, 32));
      const float mnew = fmaxf(mr[r], mx);
      const float corr = __expf(mr[r] - mnew);
      const float p0 = __expf(a0 - mnew);
      const float p1 = __expf(a1 - mnew);
      float ps = p0 + p1;
      ps += __shfl_xor(ps, 1, 32);
      ps += __shfl_xor(ps, 2, 32);
      ps += __shfl_xor(ps, 4, 32);
      ps += __shfl_xor(ps, 8, 32);
      lr[r] = lr[r] * corr + ps;
      mr[r] = mnew;
#pragma unroll
      for (int c = 0; c < 8; ++c) acc[c][r] *= corr;
      pt[r + hf * 8][l16]      = (bf16_t)p0;
      pt[r + hf * 8][16 + l16] = (bf16_t)p1;
    }

    // cross-lane LDS write->read within the wave: drain DS counter (CDNA5)
    asm volatile("s_wait_dscnt 0" ::: "memory");

    // ---- P (16x32) x V (32x128): 8 WMMAs, V fragments straight from global
    v16bf pf = cat8(*(const v8bf*)(&pt[l16][hf * 16]),
                    *(const v8bf*)(&pt[l16][hf * 16 + 8]));
#pragma unroll
    for (int c = 0; c < 8; ++c) {
      // B fragment: lane = output column d = c*16+l16; rows = keys
      // kt+hf*16 .. +15 -> contiguous in transposed V (32B aligned).
      v16bf vf = *(const v16bf*)(Vt + (size_t)(c * 16 + l16) * kS + kt + hf * 16);
      acc[c] = wmma_bf16(pf, vf, acc[c]);
    }
  }

  // ---- epilogue: out[b, row, h*128 + d] = acc / l ----
#pragma unroll
  for (int c = 0; c < 8; ++c) {
#pragma unroll
    for (int r = 0; r < 8; ++r) {
      const int row = qlo + r + hf * 8;
      ao[((size_t)b * kS + row) * (kNH * kVhd) + h * kVhd + c * 16 + l16] =
          (bf16_t)(acc[c][r] / lr[r]);
    }
  }
}

// ---------------------------------------------------------------------------
// launch
// ---------------------------------------------------------------------------
extern "C" void kernel_launch(void* const* d_in, const int* in_sizes, int n_in,
                              void* d_out, int out_size, void* d_ws, size_t ws_size,
                              hipStream_t stream) {
  (void)in_sizes; (void)n_in; (void)out_size; (void)ws_size;
  const float* x     = (const float*)d_in[0];
  const float* fcos  = (const float*)d_in[1];
  const float* fsin  = (const float*)d_in[2];
  const float* wqa   = (const float*)d_in[3];
  const float* wqab  = (const float*)d_in[4];
  const float* qnw   = (const float*)d_in[5];
  const float* wqb   = (const float*)d_in[6];
  const float* wqbb  = (const float*)d_in[7];
  const float* wkva  = (const float*)d_in[8];
  const float* wkvab = (const float*)d_in[9];
  const float* kvnw  = (const float*)d_in[10];
  const float* wkvb  = (const float*)d_in[11];
  const float* wkvbb = (const float*)d_in[12];
  const float* wo    = (const float*)d_in[13];
  const float* wob   = (const float*)d_in[14];
  float* out = (float*)d_out;

  char* w = (char*)d_ws;
  size_t off = 0;
  auto alloc = [&](size_t bytes) -> char* {
    off = (off + 255) & ~(size_t)255;
    char* p = w + off;
    off += bytes;
    return p;
  };
  bf16_t* xb   = (bf16_t*)alloc((size_t)kM * kDim * 2);
  bf16_t* wqaB = (bf16_t*)alloc((size_t)kQL * kDim * 2);
  bf16_t* wqbB = (bf16_t*)alloc((size_t)(kNH * kQkHd) * kQL * 2);
  bf16_t* wkaB = (bf16_t*)alloc((size_t)(kKVL + kRope) * kDim * 2);
  bf16_t* wkbB = (bf16_t*)alloc((size_t)(kNH * 256) * kKVL * 2);
  bf16_t* woB  = (bf16_t*)alloc((size_t)kDim * (kNH * kVhd) * 2);
  bf16_t* qa   = (bf16_t*)alloc((size_t)kM * kQL * 2);
  bf16_t* kva  = (bf16_t*)alloc((size_t)kM * (kKVL + kRope) * 2);
  bf16_t* qan  = (bf16_t*)alloc((size_t)kM * kQL * 2);
  bf16_t* kvn  = (bf16_t*)alloc((size_t)kM * kKVL * 2);
  bf16_t* qful = (bf16_t*)alloc((size_t)kM * (kNH * kQkHd) * 2);
  bf16_t* kvbO = (bf16_t*)alloc((size_t)kM * (kNH * 256) * 2);
  bf16_t* qhh  = (bf16_t*)alloc((size_t)kB * kNH * kS * kQkHd * 2);
  bf16_t* khh  = (bf16_t*)alloc((size_t)kB * kNH * kS * kQkHd * 2);
  // +256B pad: slack only; V fragment loads end at key kt+31 <= S-1, inside the
  // last row.
  bf16_t* vht  = (bf16_t*)alloc((size_t)kB * kNH * kVhd * kS * 2 + 256);
  bf16_t* ao   = (bf16_t*)alloc((size_t)kM * (kNH * kVhd) * 2);

  auto cvt = [&](const float* src, bf16_t* dst, size_t n) {
    k_cvt_bf16<<<dim3((unsigned)((n / 4 + 255) / 256)), dim3(256), 0, stream>>>(
        src, dst, (int)n);
  };
  auto gemm = [&](const bf16_t* A, const bf16_t* W, const float* bias,
                  float* of, bf16_t* ob, int M, int N, int K) {
    k_gemm_bf16_nt<<<dim3(N / 64, M / 128), dim3(128), 0, stream>>>(
        A, W, bias, of, ob, M, N, K);
  };

  // 1. fp32 -> bf16 for activations and weights
  cvt(x,    xb,   (size_t)kM * kDim);
  cvt(wqa,  wqaB, (size_t)kQL * kDim);
  cvt(wqb,  wqbB, (size_t)(kNH * kQkHd) * kQL);
  cvt(wkva, wkaB, (size_t)(kKVL + kRope) * kDim);
  cvt(wkvb, wkbB, (size_t)(kNH * 256) * kKVL);
  cvt(wo,   woB,  (size_t)kDim * (kNH * kVhd));

  // 2. down-projections
  gemm(xb, wqaB, wqab,  nullptr, qa,  kM, kQL,          kDim);
  gemm(xb, wkaB, wkvab, nullptr, kva, kM, kKVL + kRope, kDim);

  // 3. RMSNorm (q_a over 1536; kv_c = first 512 of kv_a)
  k_rmsnorm<<<dim3(kM), dim3(256), 0, stream>>>(qa,  qnw,  qan, kQL,  kQL);
  k_rmsnorm<<<dim3(kM), dim3(256), 0, stream>>>(kva, kvnw, kvn, kKVL, kKVL + kRope);

  // 4. up-projections
  gemm(qan, wqbB, wqbb,  nullptr, qful, kM, kNH * kQkHd, kQL);
  gemm(kvn, wkbB, wkvbb, nullptr, kvbO, kM, kNH * 256,   kKVL);

  // 5. RoPE + head-major repack (V transposed)
  k_pack_rope<<<dim3(kM), dim3(256), 0, stream>>>(qful, kva, kvbO, fcos, fsin,
                                                  qhh, khh, vht);

  // 6. causal flash attention (WMMA)
  k_mla_attn<<<dim3(kS / 16, kNH, kB), dim3(32), 0, stream>>>(qhh, khh, vht, ao);

  // 7. output projection (f32 result)
  gemm(ao, woB, wob, out, nullptr, kM, kDim, kNH * kVhd);
}
